// AMGSRN_77129022702001
// MI455X (gfx1250) — matrix operand order repaired and checked by reading, not compiled, not run on hardware
//
#include <hip/hip_runtime.h>
#include <hip/hip_bf16.h>

// AMGSRN fused inference for MI455X (gfx1250, wave32).
// One kernel: transform -> trilinear gather (L2-resident 32MB grids) -> f16
// feats in LDS -> MLP via v_wmma_f32_16x16x32_f16 -> f32 output.

typedef __attribute__((ext_vector_type(16))) _Float16 v16h;
typedef __attribute__((ext_vector_type(8)))  float    v8f;

#define NGRIDS 16
#define GRES   64
#define GPLANE (GRES * GRES * GRES)   // 262144 elems per channel plane
#define GSTRIDE (2 * GPLANE)          // per-grid stride (2 channels)
#define PTS_PER_BLOCK 256

union F16Frag { v16h v; uint4 q[2]; };

__global__ __launch_bounds__(256)
void amgsrn_fused(const float* __restrict__ xin,
                  const float* __restrict__ rot,
                  const float* __restrict__ scl,
                  const float* __restrict__ trn,
                  const float* __restrict__ fg,
                  const float* __restrict__ W0,
                  const float* __restrict__ b0,
                  const float* __restrict__ W1,
                  const float* __restrict__ b1,
                  const float* __restrict__ W2,
                  const float* __restrict__ b2,
                  float* __restrict__ out,
                  int npts)
{
    __shared__ alignas(16) _Float16 featsLds[PTS_PER_BLOCK * 32]; // [p][k]   16KB
    __shared__ alignas(16) _Float16 h0Lds[PTS_PER_BLOCK * 64];    // [p][n]   32KB
    __shared__ alignas(16) _Float16 w0t[64 * 32];                 // [n][k]    4KB
    __shared__ alignas(16) _Float16 w1t[64 * 64];                 // [n][k]    8KB
    __shared__ float bias0[64];
    __shared__ float bias1[64];
    __shared__ float w2s[64];
    __shared__ float Tm[NGRIDS][12];

    const int tid = threadIdx.x;

    // ---- Stage weights (transposed, f16) + biases + transforms into LDS ----
    for (int i = tid; i < 32 * 64; i += 256) {
        int k = i >> 6, n = i & 63;
        w0t[n * 32 + k] = (_Float16)W0[i];          // W0 is [k=32][n=64]
    }
    for (int i = tid; i < 64 * 64; i += 256) {
        int k = i >> 6, n = i & 63;
        w1t[n * 64 + k] = (_Float16)W1[i];          // W1 is [k=64][n=64]
    }
    if (tid < 64) { bias0[tid] = b0[tid]; bias1[tid] = b1[tid]; w2s[tid] = W2[tid]; }
    if (tid < NGRIDS) {
        const int g = tid;
        float qx = rot[g*4+0], qy = rot[g*4+1], qz = rot[g*4+2], qw = rot[g*4+3];
        float inv = rsqrtf(qx*qx + qy*qy + qz*qz + qw*qw);
        qx *= inv; qy *= inv; qz *= inv; qw *= inv;
        float sx = __expf(scl[g*3+0]), sy = __expf(scl[g*3+1]), sz = __expf(scl[g*3+2]);
        Tm[g][0] = sx * (1.f - 2.f*(qy*qy + qz*qz));
        Tm[g][1] = sx * 2.f * (qx*qy - qz*qw);
        Tm[g][2] = sx * 2.f * (qx*qz + qy*qw);
        Tm[g][3] = sy * 2.f * (qx*qy + qz*qw);
        Tm[g][4] = sy * (1.f - 2.f*(qx*qx + qz*qz));
        Tm[g][5] = sy * 2.f * (qy*qz - qx*qw);
        Tm[g][6] = sz * 2.f * (qx*qz - qy*qw);
        Tm[g][7] = sz * 2.f * (qy*qz + qx*qw);
        Tm[g][8] = sz * (1.f - 2.f*(qx*qx + qy*qy));
        Tm[g][9]  = trn[g*3+0];
        Tm[g][10] = trn[g*3+1];
        Tm[g][11] = trn[g*3+2];
    }
    __syncthreads();

    // ---- Per-point transform + trilinear gather (grids live in L2) ----
    const int pid = blockIdx.x * PTS_PER_BLOCK + tid;
    const int pc  = pid < npts ? pid : (npts - 1);
    const float X = xin[pc*3+0], Y = xin[pc*3+1], Z = xin[pc*3+2];

    for (int g = 0; g < NGRIDS; ++g) {
        const float* T = Tm[g];
        const float px = fmaf(T[0], X, fmaf(T[1], Y, fmaf(T[2], Z, T[9])));
        const float py = fmaf(T[3], X, fmaf(T[4], Y, fmaf(T[5], Z, T[10])));
        const float pz = fmaf(T[6], X, fmaf(T[7], Y, fmaf(T[8], Z, T[11])));
        const float fx = (px + 1.f) * (0.5f * 63.f);
        const float fy = (py + 1.f) * (0.5f * 63.f);
        const float fz = (pz + 1.f) * (0.5f * 63.f);
        const float x0f = floorf(fx), y0f = floorf(fy), z0f = floorf(fz);
        const int   x0 = (int)x0f,   y0 = (int)y0f,   z0 = (int)z0f;
        const float wx = fx - x0f,   wy = fy - y0f,   wz = fz - z0f;

        // Per-axis clamped indices + validity-masked weights (matches ref
        // semantics: weight zeroed if any corner coordinate out of range).
        const int xc[2] = { min(max(x0, 0), 63),     min(max(x0 + 1, 0), 63) };
        const int yc[2] = { min(max(y0, 0), 63),     min(max(y0 + 1, 0), 63) };
        const int zc[2] = { min(max(z0, 0), 63),     min(max(z0 + 1, 0), 63) };
        const float wxv[2] = { (x0 >= 0 && x0 < 64)         ? (1.f - wx) : 0.f,
                               (x0 + 1 >= 0 && x0 + 1 < 64) ? wx         : 0.f };
        const float wyv[2] = { (y0 >= 0 && y0 < 64)         ? (1.f - wy) : 0.f,
                               (y0 + 1 >= 0 && y0 + 1 < 64) ? wy         : 0.f };
        const float wzv[2] = { (z0 >= 0 && z0 < 64)         ? (1.f - wz) : 0.f,
                               (z0 + 1 >= 0 && z0 + 1 < 64) ? wz         : 0.f };

        const float* g0 = fg + (size_t)g * GSTRIDE;
        const float* g1 = g0 + GPLANE;
        float acc0 = 0.f, acc1 = 0.f;
        #pragma unroll
        for (int dz = 0; dz < 2; ++dz) {
            #pragma unroll
            for (int dy = 0; dy < 2; ++dy) {
                const int   byz = zc[dz] * 4096 + yc[dy] * 64;
                const float wyz = wzv[dz] * wyv[dy];
                #pragma unroll
                for (int dx = 0; dx < 2; ++dx) {
                    const float w = wyz * wxv[dx];
                    const int idx = byz + xc[dx];
                    acc0 = fmaf(g0[idx], w, acc0);
                    acc1 = fmaf(g1[idx], w, acc1);
                }
            }
        }
        featsLds[tid * 32 + 2 * g + 0] = (_Float16)acc0;
        featsLds[tid * 32 + 2 * g + 1] = (_Float16)acc1;
    }
    __syncthreads();

    // ---- MLP via WMMA: each wave owns its 32 points (2 M-tiles of 16) ----
    const int lane  = tid & 31;
    const int wbase = (tid >> 5) * 32;
    const int ln    = lane & 15;      // N (or M-row) index within tile
    const int hi    = lane >> 4;      // lane-half selects K sub-range
    const float bias2 = b2[0];

    #pragma unroll
    for (int mt = 0; mt < 2; ++mt) {
        const int prow = wbase + mt * 16;

        // Layer0 A fragment: 16x32 f16, rows = prow+ln.
        // Lane half hi: K halves [8hi,8hi+8) then [16+8hi,16+8hi+8).
        F16Frag a0;
        {
            const _Float16* src = &featsLds[(prow + ln) * 32];
            a0.q[0] = *(const uint4*)(src + 8 * hi);
            a0.q[1] = *(const uint4*)(src + 16 + 8 * hi);
        }

        // Layer0: feats(16x32) @ W0(32x64) + b0, ReLU -> h0 LDS
        #pragma unroll
        for (int nt = 0; nt < 4; ++nt) {
            F16Frag bf;
            {
                const _Float16* src = &w0t[(nt * 16 + ln) * 32];
                bf.q[0] = *(const uint4*)(src + 8 * hi);
                bf.q[1] = *(const uint4*)(src + 16 + 8 * hi);
            }
            v8f c;
            const float bv = bias0[nt * 16 + ln];
            #pragma unroll
            for (int r = 0; r < 8; ++r) c[r] = bv;       // bias broadcast along M
            c = __builtin_amdgcn_wmma_f32_16x16x32_f16(
                    false, a0.v, false, bf.v, (short)0, c, false, false);
            #pragma unroll
            for (int r = 0; r < 8; ++r) {
                const float h = fmaxf(c[r], 0.f);
                h0Lds[(prow + r + 8 * hi) * 64 + nt * 16 + ln] = (_Float16)h;
            }
        }

        // Wave-local LDS RAW: make h0 stores visible before A-frag reloads.
        asm volatile("s_wait_dscnt 0" ::: "memory");

        // Layer1 A fragments: K=64 in two 32-wide chunks.
        F16Frag a10, a11;
        {
            const _Float16* src = &h0Lds[(prow + ln) * 64];
            a10.q[0] = *(const uint4*)(src + 8 * hi);
            a10.q[1] = *(const uint4*)(src + 16 + 8 * hi);
            a11.q[0] = *(const uint4*)(src + 32 + 8 * hi);
            a11.q[1] = *(const uint4*)(src + 48 + 8 * hi);
        }

        float acc[8];
        #pragma unroll
        for (int r = 0; r < 8; ++r) acc[r] = 0.f;

        // Layer1 + fused layer2 partial dot (h1 never leaves registers).
        #pragma unroll
        for (int nt = 0; nt < 4; ++nt) {
            F16Frag bk0, bk1;
            {
                const _Float16* src = &w1t[(nt * 16 + ln) * 64];
                bk0.q[0] = *(const uint4*)(src + 8 * hi);
                bk0.q[1] = *(const uint4*)(src + 16 + 8 * hi);
                bk1.q[0] = *(const uint4*)(src + 32 + 8 * hi);
                bk1.q[1] = *(const uint4*)(src + 48 + 8 * hi);
            }
            v8f c;
            const float bv = bias1[nt * 16 + ln];
            #pragma unroll
            for (int r = 0; r < 8; ++r) c[r] = bv;
            c = __builtin_amdgcn_wmma_f32_16x16x32_f16(
                    false, a10.v, false, bk0.v, (short)0, c, false, false);
            c = __builtin_amdgcn_wmma_f32_16x16x32_f16(
                    false, a11.v, false, bk1.v, (short)0, c, false, false);
            const float w2v = w2s[nt * 16 + ln];
            #pragma unroll
            for (int r = 0; r < 8; ++r)
                acc[r] = fmaf(fmaxf(c[r], 0.f), w2v, acc[r]);
        }

        // Layer2 reduce over N: butterfly within each 16-lane half.
        #pragma unroll
        for (int r = 0; r < 8; ++r) {
            float v = acc[r];
            v += __shfl_xor(v, 1, 32);
            v += __shfl_xor(v, 2, 32);
            v += __shfl_xor(v, 4, 32);
            v += __shfl_xor(v, 8, 32);
            acc[r] = v + bias2;   // *(DATA_MAX-DATA_MIN)+DATA_MIN == identity
        }
        if (ln == 0) {
            const int gbase = blockIdx.x * PTS_PER_BLOCK + prow + 8 * hi;
            if (gbase + 7 < npts) {
                float4 o0; o0.x = acc[0]; o0.y = acc[1]; o0.z = acc[2]; o0.w = acc[3];
                float4 o1; o1.x = acc[4]; o1.y = acc[5]; o1.z = acc[6]; o1.w = acc[7];
                *(float4*)(out + gbase)     = o0;
                *(float4*)(out + gbase + 4) = o1;
            } else {
                #pragma unroll
                for (int r = 0; r < 8; ++r)
                    if (gbase + r < npts) out[gbase + r] = acc[r];
            }
        }
    }
}

extern "C" void kernel_launch(void* const* d_in, const int* in_sizes, int n_in,
                              void* d_out, int out_size, void* d_ws, size_t ws_size,
                              hipStream_t stream) {
    const float* x   = (const float*)d_in[0];
    const float* rot = (const float*)d_in[1];
    const float* scl = (const float*)d_in[2];
    const float* trn = (const float*)d_in[3];
    const float* fg  = (const float*)d_in[4];
    const float* W0  = (const float*)d_in[5];
    const float* b0  = (const float*)d_in[6];
    const float* W1  = (const float*)d_in[7];
    const float* b1  = (const float*)d_in[8];
    const float* W2  = (const float*)d_in[9];
    const float* b2  = (const float*)d_in[10];
    float* out = (float*)d_out;

    const int npts   = in_sizes[0] / 3;
    const int blocks = (npts + PTS_PER_BLOCK - 1) / PTS_PER_BLOCK;
    hipLaunchKernelGGL(amgsrn_fused, dim3(blocks), dim3(256), 0, stream,
                       x, rot, scl, trn, fg, W0, b0, W1, b1, W2, b2, out, npts);
}
